// BoundaryConvLayer_70918499992031
// MI455X (gfx1250) — compile-verified, matrix operand order checked by reading, not run on hardware
//
#include <hip/hip_runtime.h>
#include <hip/hip_bf16.h>
#include <math.h>

typedef __attribute__((ext_vector_type(16))) _Float16 v16h;
typedef __attribute__((ext_vector_type(8)))  float    v8f;

#define NROWS 50000
#define NP    50048           // padded to multiple of 64 (4 waves * 16 rows)
#define HD    128
#define NEDGE 800000

// ---------- helpers ----------------------------------------------------------

__device__ __forceinline__ float softplusf(float v) {
  return v > 20.f ? v : log1pf(expf(v));
}
__device__ __forceinline__ float geluf(float v) {
  return 0.5f * v * (1.f + erff(v * 0.70710678118654752f));
}

// Load a 16x32 f16 A-tile (rows m0..m0+15, K = kt*32..kt*32+31) from a
// row-major f16 matrix with leading dim 128.
// Lane l<16: row=l,  halfs j<8 -> K=kt*32+j, j>=8 -> K=kt*32+16+(j-8)
// Lane l>=16: row=l-16, same but K offset +8.
__device__ __forceinline__ v16h load_a16(const _Float16* base, int lane, int kt) {
  int row = lane & 15;
  int k0  = kt * 32 + ((lane & 16) ? 8 : 0);
  const _Float16* p = base + row * HD + k0;
  union { v16h v; float4 f[2]; } u;
  u.f[0] = *(const float4*)(p);        // halfs 0..7
  u.f[1] = *(const float4*)(p + 16);   // halfs 8..15
  return u.v;
}

// Load a pre-packed 32x16 B-tile: packed layout puts each lane's 16 halfs
// contiguously:  offset = ((nt*4+kt)*32 + lane)*16
__device__ __forceinline__ v16h load_b16(const _Float16* wp, int lane, int kt, int nt) {
  const _Float16* p = wp + ((((nt << 2) + kt) << 5) + lane) * 16;
  union { v16h v; float4 f[2]; } u;
  u.f[0] = *(const float4*)(p);
  u.f[1] = *(const float4*)(p + 8);
  return u.v;
}

// 16x128 = (16x128) @ (128x128) + bias, f16 in / f32 C-layout out.
// c[nt][r] holds C[row = r + (lane>=16?8:0)][col = nt*16 + (lane&15)]
__device__ __forceinline__ void mm128(const v16h a[4], const _Float16* wp,
                                      const float* bias, int lane, v8f c[8]) {
  int col = lane & 15;
#pragma unroll
  for (int nt = 0; nt < 8; ++nt) {
    v8f acc = {};
#pragma unroll
    for (int kt = 0; kt < 4; ++kt) {
      v16h b = load_b16(wp, lane, kt, nt);
      acc = __builtin_amdgcn_wmma_f32_16x16x32_f16(
          false, a[kt], false, b, (short)0, acc, false, false);
    }
    float bv = bias[nt * 16 + col];
#pragma unroll
    for (int r = 0; r < 8; ++r) acc[r] += bv;
    c[nt] = acc;
  }
}

// Store C-layout tiles as f16 row-major into a 16x128 LDS scratch, with an
// optional activation (0=none, 1=relu, 2=gelu). Then wait for DS stores.
__device__ __forceinline__ void c_to_lds(_Float16* lds, const v8f c[8],
                                         int lane, int act) {
  int col = lane & 15;
  int rb  = (lane & 16) ? 8 : 0;
#pragma unroll
  for (int nt = 0; nt < 8; ++nt) {
#pragma unroll
    for (int r = 0; r < 8; ++r) {
      float v = c[nt][r];
      if (act == 1)      v = fmaxf(v, 0.f);
      else if (act == 2) v = geluf(v);
      lds[(rb + r) * HD + nt * 16 + col] = (_Float16)v;
    }
  }
  asm volatile("s_wait_dscnt 0" ::: "memory");
}

// ---------- kernel 0: pack one 128x128 f32 weight into WMMA-B f16 layout ----
__global__ __launch_bounds__(256) void pack_w(const float* __restrict__ w,
                                              _Float16* __restrict__ dst) {
  int t = blockIdx.x * 256 + threadIdx.x;      // 16384 elements
  if (t >= 128 * 128) return;
  int j    = t & 15;
  int lane = (t >> 4) & 31;
  int kt   = (t >> 9) & 3;
  int nt   = t >> 11;
  int n    = nt * 16 + (lane & 15);
  int khi  = (lane & 16) ? 8 : 0;
  int k    = kt * 32 + khi + (j < 8 ? j : j + 8);
  dst[t] = (_Float16)w[k * HD + n];
}

// ---------- kernel 1: h = x @ W_lin + b (f32 + f16 copies) ------------------
__global__ __launch_bounds__(128) void h_gemm(const float* __restrict__ x,
                                              const _Float16* __restrict__ wlin,
                                              const float* __restrict__ blin,
                                              float* __restrict__ h32,
                                              _Float16* __restrict__ h16) {
  int lane = threadIdx.x & 31;
  int wid  = threadIdx.x >> 5;
  int m0   = (blockIdx.x * 4 + wid) * 16;

  v16h a[4];
#pragma unroll
  for (int kt = 0; kt < 4; ++kt) {
    int row = m0 + (lane & 15);
    if (row >= NROWS) row = NROWS - 1;             // clamp (padded rows unused)
    int k0 = kt * 32 + ((lane & 16) ? 8 : 0);
    const float* p = x + (size_t)row * HD + k0;
    v16h av;
#pragma unroll
    for (int jj = 0; jj < 8; ++jj) { av[jj] = (_Float16)p[jj]; av[jj + 8] = (_Float16)p[16 + jj]; }
    a[kt] = av;
  }

  v8f c[8];
  mm128(a, wlin, blin, lane, c);

  int col = lane & 15;
  int rb  = (lane & 16) ? 8 : 0;
#pragma unroll
  for (int nt = 0; nt < 8; ++nt) {
#pragma unroll
    for (int r = 0; r < 8; ++r) {
      int m = m0 + rb + r;
      int cc = nt * 16 + col;
      float v = c[nt][r];
      h32[(size_t)m * HD + cc] = v;
      h16[(size_t)m * HD + cc] = (_Float16)v;
    }
  }
}

// ---------- kernel 2: zero agg ----------------------------------------------
__global__ __launch_bounds__(256) void zero_k(float* __restrict__ p, int n) {
  int t = blockIdx.x * 256 + threadIdx.x;
  if (t < n) p[t] = 0.f;
}

// ---------- kernel 3: agg[dst] += h[src] over all edges ---------------------
__global__ __launch_bounds__(256) void scatter_add(const long long* __restrict__ ei,
                                                   const float* __restrict__ h32,
                                                   float* __restrict__ agg) {
  int t = blockIdx.x * 256 + threadIdx.x;
  int e = t >> 7;            // 128 columns per edge
  int c = t & 127;
  if (e < NEDGE) {
    int src = (int)ei[e];
    int dst = (int)ei[NEDGE + e];
    unsafeAtomicAdd(&agg[(size_t)dst * HD + c], h32[(size_t)src * HD + c]);
  }
}

// ---------- kernel 4: x_res = layernorm(h) ----------------------------------
__global__ __launch_bounds__(128) void layernorm_k(const float* __restrict__ h32,
                                                   const float* __restrict__ g,
                                                   const float* __restrict__ b,
                                                   float* __restrict__ xres) {
  __shared__ float red[128];
  int row = blockIdx.x;
  int tid = threadIdx.x;
  float v = h32[(size_t)row * HD + tid];
  red[tid] = v;
  __syncthreads();
#pragma unroll
  for (int s = 64; s > 0; s >>= 1) { if (tid < s) red[tid] += red[tid + s]; __syncthreads(); }
  float mean = red[0] * (1.f / 128.f);
  __syncthreads();
  float d = v - mean;
  red[tid] = d * d;
  __syncthreads();
#pragma unroll
  for (int s = 64; s > 0; s >>= 1) { if (tid < s) red[tid] += red[tid + s]; __syncthreads(); }
  float var = red[0] * (1.f / 128.f);
  xres[(size_t)row * HD + tid] = d * rsqrtf(var + 1e-5f) * g[tid] + b[tid];
}

// ---------- kernel 5: fused gate MLPs + message norm + output MLP -----------
// packed weights at wp + i*16384: [0]=Wlin (unused here) [1]=Wa1 [2]=Wa2
// [3]=Wb1 [4]=Wb2 [5]=Wg1 [6]=Wg2 [7]=Wf1 [8]=Wf2
__global__ __launch_bounds__(128) void fused_epilogue(
    const _Float16* __restrict__ h16, const float* __restrict__ agg,
    const float* __restrict__ xres, const float* __restrict__ degree,
    const _Float16* __restrict__ wp,
    const float* __restrict__ ba1, const float* __restrict__ ba2,
    const float* __restrict__ bb1, const float* __restrict__ bb2,
    const float* __restrict__ bg1, const float* __restrict__ bg2,
    const float* __restrict__ bf1, const float* __restrict__ bf2,
    float* __restrict__ out) {
  __shared__ _Float16 scr[4][16 * HD];
  int lane = threadIdx.x & 31;
  int wid  = threadIdx.x >> 5;
  int m0   = (blockIdx.x * 4 + wid) * 16;
  _Float16* lds = scr[wid];

  v16h Ah[4], At[4];
#pragma unroll
  for (int kt = 0; kt < 4; ++kt) Ah[kt] = load_a16(h16 + (size_t)m0 * HD, lane, kt);

  v8f t[8], alpha[8], beta[8], gam[8];

  // alpha = softplus( relu(h@Wa1+ba1) @ Wa2 + ba2 )
  mm128(Ah, wp + 1 * 16384, ba1, lane, t);
  c_to_lds(lds, t, lane, /*relu*/1);
#pragma unroll
  for (int kt = 0; kt < 4; ++kt) At[kt] = load_a16(lds, lane, kt);
  mm128(At, wp + 2 * 16384, ba2, lane, alpha);
#pragma unroll
  for (int nt = 0; nt < 8; ++nt)
#pragma unroll
    for (int r = 0; r < 8; ++r) alpha[nt][r] = softplusf(alpha[nt][r]);

  // beta = softplus( relu(h@Wb1+bb1) @ Wb2 + bb2 )
  mm128(Ah, wp + 3 * 16384, bb1, lane, t);
  c_to_lds(lds, t, lane, 1);
#pragma unroll
  for (int kt = 0; kt < 4; ++kt) At[kt] = load_a16(lds, lane, kt);
  mm128(At, wp + 4 * 16384, bb2, lane, beta);
#pragma unroll
  for (int nt = 0; nt < 8; ++nt)
#pragma unroll
    for (int r = 0; r < 8; ++r) beta[nt][r] = softplusf(beta[nt][r]);

  // gamma = gelu(h@Wg1+bg1) @ Wg2 + bg2
  mm128(Ah, wp + 5 * 16384, bg1, lane, t);
  c_to_lds(lds, t, lane, /*gelu*/2);
#pragma unroll
  for (int kt = 0; kt < 4; ++kt) At[kt] = load_a16(lds, lane, kt);
  mm128(At, wp + 6 * 16384, bg2, lane, gam);

  // y = (beta*agg + gamma) / (alpha + beta*degree)  -> LDS as f16 A operand
  int col = lane & 15;
  int rb  = (lane & 16) ? 8 : 0;
#pragma unroll
  for (int nt = 0; nt < 8; ++nt) {
#pragma unroll
    for (int r = 0; r < 8; ++r) {
      int m = m0 + rb + r;
      int c = nt * 16 + col;
      float av = agg[(size_t)m * HD + c];
      float dg = (m < NROWS) ? degree[m] : 0.f;
      float bt = beta[nt][r];
      float y  = (bt * av + gam[nt][r]) / (alpha[nt][r] + bt * dg);
      lds[(rb + r) * HD + c] = (_Float16)y;
    }
  }
  asm volatile("s_wait_dscnt 0" ::: "memory");
#pragma unroll
  for (int kt = 0; kt < 4; ++kt) At[kt] = load_a16(lds, lane, kt);

  // z = gelu(y@Wf1+bf1) @ Wf2 + bf2 ; out = z + x_res
  mm128(At, wp + 7 * 16384, bf1, lane, t);
  c_to_lds(lds, t, lane, 2);
#pragma unroll
  for (int kt = 0; kt < 4; ++kt) At[kt] = load_a16(lds, lane, kt);
  mm128(At, wp + 8 * 16384, bf2, lane, t);

#pragma unroll
  for (int nt = 0; nt < 8; ++nt) {
#pragma unroll
    for (int r = 0; r < 8; ++r) {
      int m = m0 + rb + r;
      if (m < NROWS) {
        int c = nt * 16 + col;
        out[(size_t)m * HD + c] = t[nt][r] + xres[(size_t)m * HD + c];
      }
    }
  }
}

// ---------- host ------------------------------------------------------------
extern "C" void kernel_launch(void* const* d_in, const int* in_sizes, int n_in,
                              void* d_out, int out_size, void* d_ws, size_t ws_size,
                              hipStream_t stream) {
  (void)in_sizes; (void)n_in; (void)out_size; (void)ws_size;

  const float*     x      = (const float*)d_in[0];
  const long long* ei     = (const long long*)d_in[1];
  const float*     degree = (const float*)d_in[2];
  const float* W[9] = {
      (const float*)d_in[3],  (const float*)d_in[5],  (const float*)d_in[7],
      (const float*)d_in[9],  (const float*)d_in[11], (const float*)d_in[13],
      (const float*)d_in[15], (const float*)d_in[17], (const float*)d_in[19]};
  const float* blin = (const float*)d_in[4];
  const float* ba1  = (const float*)d_in[6];
  const float* ba2  = (const float*)d_in[8];
  const float* bb1  = (const float*)d_in[10];
  const float* bb2  = (const float*)d_in[12];
  const float* bg1  = (const float*)d_in[14];
  const float* bg2  = (const float*)d_in[16];
  const float* bf1  = (const float*)d_in[18];
  const float* bf2  = (const float*)d_in[20];
  const float* ln_g = (const float*)d_in[21];
  const float* ln_b = (const float*)d_in[22];

  char* ws = (char*)d_ws;
  size_t o = 0;
  float*     h32   = (float*)(ws + o);     o += (size_t)NP * HD * 4;
  _Float16*  h16   = (_Float16*)(ws + o);  o += (size_t)NP * HD * 2;
  float*     agg   = (float*)(ws + o);     o += (size_t)NP * HD * 4;
  float*     xres  = (float*)(ws + o);     o += (size_t)NP * HD * 4;
  _Float16*  wpack = (_Float16*)(ws + o);  o += (size_t)9 * 16384 * 2;

  for (int i = 0; i < 9; ++i)
    pack_w<<<64, 256, 0, stream>>>(W[i], wpack + (size_t)i * 16384);

  h_gemm<<<NP / 64, 128, 0, stream>>>(x, wpack, blin, h32, h16);

  zero_k<<<(NP * HD + 255) / 256, 256, 0, stream>>>(agg, NP * HD);

  scatter_add<<<(NEDGE * 128) / 256, 256, 0, stream>>>(ei, h32, agg);

  layernorm_k<<<NROWS, 128, 0, stream>>>(h32, ln_g, ln_b, xres);

  fused_epilogue<<<NP / 64, 128, 0, stream>>>(
      h16, agg, xres, degree, wpack,
      ba1, ba2, bb1, bb2, bg1, bg2, bf1, bf2, (float*)d_out);
}
